// OptimizedMoEAttention_57690000720286
// MI455X (gfx1250) — matrix operand order repaired and checked by reading, not compiled, not run on hardware
//
#include <hip/hip_runtime.h>
#include <hip/hip_bf16.h>
#include <math.h>

#define B_  2
#define S_  1024
#define E_  1024
#define NE_ 8
#define H_  16
#define DH_ 64
#define M_  (B_ * S_)   /* 2048 tokens */
#define EH_ (E_ / 2)    /* 512 router hidden */

typedef __attribute__((ext_vector_type(16))) _Float16 v16h;
typedef __attribute__((ext_vector_type(8)))  _Float16 v8h;
typedef __attribute__((ext_vector_type(8)))  float    v8f;

__device__ __forceinline__ v8f wmma_f16(v16h a, v16h b, v8f c) {
  // emits v_wmma_f32_16x16x32_f16
  return __builtin_amdgcn_wmma_f32_16x16x32_f16(false, a, false, b, (short)0, c,
                                                false, false);
}

// ---- CDNA5 async global->LDS copy (tracked by ASYNCcnt) ---------------------
// VDST VGPR holds the LDS byte address; VADDR pair holds the 64-bit global
// address (GV mode, saddr = off).  16-byte aligned.
__device__ __forceinline__ void async_g2l_b128(unsigned lds_byte_off, const void* g) {
  asm volatile("global_load_async_to_lds_b128 %0, %1, off"
               :: "v"(lds_byte_off), "v"((unsigned long long)(uintptr_t)g)
               : "memory");
}
__device__ __forceinline__ void wait_async_zero() {
  asm volatile("s_wait_asynccnt 0x0" ::: "memory");
}

// ---- fragment loaders (CDNA5 wave32 16x16x32 f16 layouts) -------------------
// A (16x32, MxK): lane l -> row r=l&15; halves p=0..15 = K 16*(l>>4)+p.
__device__ __forceinline__ v16h load_a_f16(const _Float16* base, int ld, int lane) {
  const int r = lane & 15, hi = lane >> 4;
  const _Float16* p = base + (size_t)r * ld + hi * 16;
  v8h lo = *(const v8h*)p;
  v8h hv = *(const v8h*)(p + 8);
  v16h out;
#pragma unroll
  for (int i = 0; i < 8; ++i) { out[i] = lo[i]; out[8 + i] = hv[i]; }
  return out;
}

// B (32x16, KxN): lane l -> col n=l&15; halves p=0..15 = K 16*(l>>4)+p.
// Element [n][k] read from base[n*ld + k] (i.e. Bᵀ rows, contiguous in k).
__device__ __forceinline__ v16h load_b_f16(const _Float16* base, int ld, int lane) {
  const int n = lane & 15, hi = lane >> 4;
  return *(const v16h*)(base + (size_t)n * ld + hi * 16);
}

// ---- utility kernels --------------------------------------------------------
__global__ void zero_kernel(float* __restrict__ p, int n) {
  int i = blockIdx.x * 256 + threadIdx.x;
  if (i < n) p[i] = 0.f;
}

__global__ void cast_kernel(const float* __restrict__ x, _Float16* __restrict__ y, int n) {
  int i = blockIdx.x * 256 + threadIdx.x;
  if (i < n) y[i] = (_Float16)x[i];
}

// ---- router: Linear(1024->512) -> LN -> ReLU -> Linear(512->8) -> softmax
//      -> top-2 renormalized -> dense gates[NE][M] ---------------------------
__global__ void __launch_bounds__(256)
router_kernel(const float* __restrict__ x,
              const float* __restrict__ rW1, const float* __restrict__ rb1,
              const float* __restrict__ ln_g, const float* __restrict__ ln_b,
              const float* __restrict__ rW2, const float* __restrict__ rb2,
              float* __restrict__ gates) {
  __shared__ float xs[E_];
  __shared__ float hs[EH_];
  __shared__ float red[256];
  __shared__ float lg[NE_];
  const int t = blockIdx.x;       // token 0..M_-1
  const int tid = threadIdx.x;

  const float* xr = x + (size_t)t * E_;
  for (int i = tid; i < E_; i += 256) xs[i] = xr[i];
  __syncthreads();

  for (int j = tid; j < EH_; j += 256) {
    const float* w = rW1 + (size_t)j * E_;
    float acc = rb1[j];
    for (int k = 0; k < E_; ++k) acc = fmaf(xs[k], w[k], acc);
    hs[j] = acc;
  }
  __syncthreads();

  float s1 = 0.f, s2 = 0.f;
  for (int j = tid; j < EH_; j += 256) { float v = hs[j]; s1 += v; s2 += v * v; }
  red[tid] = s1;
  __syncthreads();
  for (int off = 128; off > 0; off >>= 1) {
    if (tid < off) red[tid] += red[tid + off];
    __syncthreads();
  }
  const float mean = red[0] * (1.0f / EH_);
  __syncthreads();
  red[tid] = s2;
  __syncthreads();
  for (int off = 128; off > 0; off >>= 1) {
    if (tid < off) red[tid] += red[tid + off];
    __syncthreads();
  }
  const float var = red[0] * (1.0f / EH_) - mean * mean;
  const float inv = rsqrtf(var + 1e-5f);
  __syncthreads();

  for (int j = tid; j < EH_; j += 256) {
    float v = (hs[j] - mean) * inv * ln_g[j] + ln_b[j];
    hs[j] = v > 0.f ? v : 0.f;
  }
  __syncthreads();

  if (tid < NE_) {
    const float* w = rW2 + (size_t)tid * EH_;
    float acc = rb2[tid];
    for (int k = 0; k < EH_; ++k) acc = fmaf(hs[k], w[k], acc);
    lg[tid] = acc;
  }
  __syncthreads();

  if (tid == 0) {
    float mx = lg[0];
    for (int e = 1; e < NE_; ++e) mx = fmaxf(mx, lg[e]);
    float p[NE_];
    float s = 0.f;
    for (int e = 0; e < NE_; ++e) { p[e] = __expf(lg[e] - mx); s += p[e]; }
    const float sinv = 1.f / s;
    for (int e = 0; e < NE_; ++e) p[e] *= sinv;
    int i1 = 0;
    for (int e = 1; e < NE_; ++e) if (p[e] > p[i1]) i1 = e;
    int i2 = (i1 == 0) ? 1 : 0;
    for (int e = 0; e < NE_; ++e) if (e != i1 && p[e] > p[i2]) i2 = e;
    const float dinv = 1.f / (p[i1] + p[i2]);
    for (int e = 0; e < NE_; ++e) {
      float g = (e == i1) ? p[i1] * dinv : (e == i2 ? p[i2] * dinv : 0.f);
      gates[(size_t)e * M_ + t] = g;
    }
  }
}

// ---- projection GEMM: dst = x @ Wᵀ + b --------------------------------------
// Block = one 32-feature slab shared by 8 waves covering 512 tokens.
// The slab (32 x 1024 f16 = 64 KB) is DMA'd to LDS once with
// global_load_async_to_lds_b128; each wave then runs a 4(m) x 2(n) register
// block, B fragments from LDS (ds_load_b128), A streamed from global.
template <int TRANS>
__global__ void __launch_bounds__(256)
proj_gemm_kernel(const _Float16* __restrict__ A, const _Float16* __restrict__ W,
                 const float* __restrict__ bias, _Float16* __restrict__ dst,
                 int dst_ld) {
  extern __shared__ _Float16 Bs[];   // [32][E_]
  const int lane = threadIdx.x & 31;
  const int wid = threadIdx.x >> 5;
  const int ngrp = blockIdx.x & 31;  // 32 feature slabs
  const int mblk = blockIdx.x >> 5;  // 4 token blocks of 512
  const int n0 = ngrp * 32;
  const int m0 = mblk * 512 + wid * 64;

  // async-stage the weight slab: 32 rows x 1024 k = 4096 x 16B chunks
  const unsigned lds_base = (unsigned)__builtin_amdgcn_groupstaticsize();
  const _Float16* wsrc = W + (size_t)n0 * E_;
  for (int i = threadIdx.x; i < 4096; i += 256)
    async_g2l_b128(lds_base + (unsigned)i * 16u, wsrc + (size_t)i * 8);
  wait_async_zero();
  __syncthreads();

  v8f acc[4][2] = {};
  for (int kk = 0; kk < E_; kk += 32) {
    const v16h b0 = load_b_f16(Bs + (size_t)0 * E_ + kk, E_, lane);
    const v16h b1 = load_b_f16(Bs + (size_t)16 * E_ + kk, E_, lane);
#pragma unroll
    for (int mi = 0; mi < 4; ++mi) {
      const v16h a = load_a_f16(A + (size_t)(m0 + 16 * mi) * E_ + kk, E_, lane);
      acc[mi][0] = wmma_f16(a, b0, acc[mi][0]);
      acc[mi][1] = wmma_f16(a, b1, acc[mi][1]);
    }
  }

  const int hi = lane >> 4;
#pragma unroll
  for (int ni = 0; ni < 2; ++ni) {
    const int n = n0 + 16 * ni + (lane & 15);
    const float bv = bias[n];
#pragma unroll
    for (int mi = 0; mi < 4; ++mi) {
#pragma unroll
      for (int i = 0; i < 8; ++i) {
        const int m = m0 + 16 * mi + i + 8 * hi;
        const _Float16 v = (_Float16)(acc[mi][ni][i] + bv);
        if (TRANS) dst[(size_t)n * dst_ld + m] = v;   // Vᵀ [E_][M_]
        else       dst[(size_t)m * dst_ld + n] = v;   // Q/K [M_][E_]
      }
    }
  }
}

// ---- flash attention per (batch, head, 16-query tile) ----------------------
// scoresᵀ = K·Qᵀ keeps the softmax axis in-lane; Pᵀ feeds outᵀ += Vᵀ·Pᵀ with
// only xor-16 shuffles.  Q,K row-major f16; V transposed f16.
__global__ void __launch_bounds__(256)
attn_kernel(const _Float16* __restrict__ Q, const _Float16* __restrict__ K,
            const _Float16* __restrict__ Vt, _Float16* __restrict__ O) {
  const int lane = threadIdx.x & 31;
  const int w = blockIdx.x * 8 + (threadIdx.x >> 5);
  const int qt = w & 63;
  const int h = (w >> 6) & 15;
  const int b = w >> 10;
  const int hi = lane >> 4;
  const bool upper = hi != 0;

  const int q0 = b * S_ + qt * 16;  // global query token base
  const int d0 = h * DH_;

  // Q as B-fragments: two 32-wide chunks covering Dh=64
  const v16h bq0 = load_b_f16(Q + (size_t)q0 * E_ + d0, E_, lane);
  const v16h bq1 = load_b_f16(Q + (size_t)q0 * E_ + d0 + 32, E_, lane);

  v8f o0 = {}, o1 = {}, o2 = {}, o3 = {};
  float m_run = -3.0e38f, l_run = 0.f;
  const float scale = 0.125f;  // 1/sqrt(64)

  for (int kc = 0; kc < S_; kc += 32) {
    const _Float16* kb0 = K + (size_t)(b * S_ + kc) * E_ + d0;
    const _Float16* kb1 = kb0 + (size_t)16 * E_;
    v8f s0 = {}, s1 = {};
    s0 = wmma_f16(load_a_f16(kb0, E_, lane), bq0, s0);
    s0 = wmma_f16(load_a_f16(kb0 + 32, E_, lane), bq1, s0);
    s1 = wmma_f16(load_a_f16(kb1, E_, lane), bq0, s1);
    s1 = wmma_f16(load_a_f16(kb1 + 32, E_, lane), bq1, s1);

#pragma unroll
    for (int i = 0; i < 8; ++i) { s0[i] *= scale; s1[i] *= scale; }

    // online softmax along k (per column q = lane&15); lane^16 holds the
    // other 16 k-values of the same column.
    float mloc = -3.0e38f;
#pragma unroll
    for (int i = 0; i < 8; ++i) mloc = fmaxf(mloc, fmaxf(s0[i], s1[i]));
    mloc = fmaxf(mloc, __shfl_xor(mloc, 16, 32));
    const float mnew = fmaxf(m_run, mloc);
    const float corr = __expf(m_run - mnew);

    v8f p0, p1;
    float lloc = 0.f;
#pragma unroll
    for (int i = 0; i < 8; ++i) {
      p0[i] = __expf(s0[i] - mnew);
      p1[i] = __expf(s1[i] - mnew);
      lloc += p0[i] + p1[i];
    }
    lloc += __shfl_xor(lloc, 16, 32);
    l_run = l_run * corr + lloc;
    m_run = mnew;
#pragma unroll
    for (int i = 0; i < 8; ++i) {
      o0[i] *= corr; o1[i] *= corr; o2[i] *= corr; o3[i] *= corr;
    }

    // Pᵀ as B-fragment: halves p -> k = 16*hi + p; other half via xor-16.
    v16h pb;
#pragma unroll
    for (int i = 0; i < 8; ++i) {
      const float sh0 = __shfl_xor(p0[i], 16, 32);
      const float sh1 = __shfl_xor(p1[i], 16, 32);
      pb[i]     = (_Float16)(upper ? sh1 : p0[i]);
      pb[8 + i] = (_Float16)(upper ? p1[i] : sh0);
    }

    const size_t vtok = (size_t)(b * S_ + kc);
    o0 = wmma_f16(load_a_f16(Vt + (size_t)(d0 + 0)  * M_ + vtok, M_, lane), pb, o0);
    o1 = wmma_f16(load_a_f16(Vt + (size_t)(d0 + 16) * M_ + vtok, M_, lane), pb, o1);
    o2 = wmma_f16(load_a_f16(Vt + (size_t)(d0 + 32) * M_ + vtok, M_, lane), pb, o2);
    o3 = wmma_f16(load_a_f16(Vt + (size_t)(d0 + 48) * M_ + vtok, M_, lane), pb, o3);
  }

  const float linv = 1.f / l_run;
  const int qg = q0 + (lane & 15);
  _Float16* orow = O + (size_t)qg * E_ + d0;
#pragma unroll
  for (int i = 0; i < 8; ++i) {
    const int dr = i + 8 * hi;
    orow[dr]      = (_Float16)(o0[i] * linv);
    orow[16 + dr] = (_Float16)(o1[i] * linv);
    orow[32 + dr] = (_Float16)(o2[i] * linv);
    orow[48 + dr] = (_Float16)(o3[i] * linv);
  }
}

// ---- output projection + gating + accumulation -----------------------------
// finalᵀ[n,q] = oW(A) · attnᵀ(B); out[q,n] += gate[q] * (acc + ob[n]).
// Same LDS staging: the block's 32 oW rows are async-DMA'd once; waves run
// a 2(n) x 4(q) register block (A from LDS, B streamed from global).
__global__ void __launch_bounds__(256)
oproj_kernel(const _Float16* __restrict__ Attn, const _Float16* __restrict__ oWh,
             const float* __restrict__ ob, const float* __restrict__ gate,
             float* __restrict__ out) {
  extern __shared__ _Float16 Ws[];   // [32][E_]
  const int lane = threadIdx.x & 31;
  const int wid = threadIdx.x >> 5;
  const int ngrp = blockIdx.x & 31;  // 32 feature slabs
  const int qblk = blockIdx.x >> 5;  // 4 token blocks of 512
  const int n0 = ngrp * 32;
  const int q0 = qblk * 512 + wid * 64;

  const unsigned lds_base = (unsigned)__builtin_amdgcn_groupstaticsize();
  const _Float16* wsrc = oWh + (size_t)n0 * E_;
  for (int i = threadIdx.x; i < 4096; i += 256)
    async_g2l_b128(lds_base + (unsigned)i * 16u, wsrc + (size_t)i * 8);
  wait_async_zero();
  __syncthreads();

  v8f acc[2][4] = {};
  for (int kk = 0; kk < E_; kk += 32) {
    const v16h a0 = load_a_f16(Ws + (size_t)0 * E_ + kk, E_, lane);
    const v16h a1 = load_a_f16(Ws + (size_t)16 * E_ + kk, E_, lane);
#pragma unroll
    for (int qi = 0; qi < 4; ++qi) {
      const v16h bfr = load_b_f16(Attn + (size_t)(q0 + 16 * qi) * E_ + kk, E_, lane);
      acc[0][qi] = wmma_f16(a0, bfr, acc[0][qi]);
      acc[1][qi] = wmma_f16(a1, bfr, acc[1][qi]);
    }
  }

  const int hi = lane >> 4;
#pragma unroll
  for (int qi = 0; qi < 4; ++qi) {
    const int q = q0 + 16 * qi + (lane & 15);
    const float g = gate[q];
#pragma unroll
    for (int ni = 0; ni < 2; ++ni) {
#pragma unroll
      for (int i = 0; i < 8; ++i) {
        const int n = n0 + 16 * ni + i + 8 * hi;
        out[(size_t)q * E_ + n] += g * (acc[ni][qi][i] + ob[n]);
      }
    }
  }
}

// ---- host-side orchestration -----------------------------------------------
extern "C" void kernel_launch(void* const* d_in, const int* in_sizes, int n_in,
                              void* d_out, int out_size, void* d_ws, size_t ws_size,
                              hipStream_t stream) {
  const float* x    = (const float*)d_in[0];
  const float* rW1  = (const float*)d_in[1];
  const float* rb1  = (const float*)d_in[2];
  const float* ln_g = (const float*)d_in[3];
  const float* ln_b = (const float*)d_in[4];
  const float* rW2  = (const float*)d_in[5];
  const float* rb2  = (const float*)d_in[6];
  const float* qW   = (const float*)d_in[7];
  const float* qb   = (const float*)d_in[8];
  const float* kW   = (const float*)d_in[9];
  const float* kb   = (const float*)d_in[10];
  const float* vW   = (const float*)d_in[11];
  const float* vb   = (const float*)d_in[12];
  const float* oW   = (const float*)d_in[13];
  const float* ob   = (const float*)d_in[14];
  float* out = (float*)d_out;

  char* ws = (char*)d_ws;
  size_t off = 0;
  auto alloc = [&](size_t bytes) {
    void* p = ws + off;
    off = (off + bytes + 255) & ~(size_t)255;
    return p;
  };
  float*    gates = (float*)alloc((size_t)NE_ * M_ * sizeof(float));
  _Float16* xh    = (_Float16*)alloc((size_t)M_ * E_ * sizeof(_Float16));
  _Float16* Qh    = (_Float16*)alloc((size_t)M_ * E_ * sizeof(_Float16));
  _Float16* Kh    = (_Float16*)alloc((size_t)M_ * E_ * sizeof(_Float16));
  _Float16* Vt    = (_Float16*)alloc((size_t)M_ * E_ * sizeof(_Float16));
  _Float16* Ah    = (_Float16*)alloc((size_t)M_ * E_ * sizeof(_Float16));
  _Float16* qWh   = (_Float16*)alloc((size_t)E_ * E_ * sizeof(_Float16));
  _Float16* kWh   = (_Float16*)alloc((size_t)E_ * E_ * sizeof(_Float16));
  _Float16* vWh   = (_Float16*)alloc((size_t)E_ * E_ * sizeof(_Float16));
  _Float16* oWh   = (_Float16*)alloc((size_t)E_ * E_ * sizeof(_Float16));

  const int ntot = M_ * E_;       // 2M activations
  const int wtot = E_ * E_;       // 1M weights per matrix
  const size_t slab = 32u * E_ * sizeof(_Float16);  // 64 KB dynamic LDS
  zero_kernel<<<(ntot + 255) / 256, 256, 0, stream>>>(out, ntot);
  cast_kernel<<<(ntot + 255) / 256, 256, 0, stream>>>(x, xh, ntot);
  router_kernel<<<M_, 256, 0, stream>>>(x, rW1, rb1, ln_g, ln_b, rW2, rb2, gates);

  for (int e = 0; e < NE_; ++e) {
    const size_t wo = (size_t)e * E_ * E_;
    const size_t bo = (size_t)e * E_;
    // stage this expert's weights to f16 once (kills in-loop cvt)
    cast_kernel<<<(wtot + 255) / 256, 256, 0, stream>>>(qW + wo, qWh, wtot);
    cast_kernel<<<(wtot + 255) / 256, 256, 0, stream>>>(kW + wo, kWh, wtot);
    cast_kernel<<<(wtot + 255) / 256, 256, 0, stream>>>(vW + wo, vWh, wtot);
    cast_kernel<<<(wtot + 255) / 256, 256, 0, stream>>>(oW + wo, oWh, wtot);

    proj_gemm_kernel<0><<<128, 256, slab, stream>>>(xh, qWh, qb + bo, Qh, E_);
    proj_gemm_kernel<0><<<128, 256, slab, stream>>>(xh, kWh, kb + bo, Kh, E_);
    proj_gemm_kernel<1><<<128, 256, slab, stream>>>(xh, vWh, vb + bo, Vt, M_);
    attn_kernel<<<256, 256, 0, stream>>>(Qh, Kh, Vt, Ah);
    oproj_kernel<<<128, 256, slab, stream>>>(Ah, oWh, ob + bo,
                                             gates + (size_t)e * M_, out);
  }

  (void)in_sizes; (void)n_in; (void)out_size; (void)ws_size;
}
